// HyenaOperator_31293131719040
// MI455X (gfx1250) — compile-verified, matrix operand order checked by reading, not run on hardware
//
#include <hip/hip_runtime.h>
#include <hip/hip_bf16.h>

// ---------------- problem constants ----------------
constexpr int Bc = 8;
constexpr int Lc = 4096;
constexpr int Dc = 1024;
constexpr int TDc = 3 * Dc;        // 3072
constexpr int FO = 64;             // FILTER_ORDER
constexpr int FFT_N = 8192;        // 2*L
constexpr int FFT_LOG = 13;

typedef __bf16 bf16_t;
typedef __attribute__((ext_vector_type(16))) __bf16 v16bf;
typedef __attribute__((ext_vector_type(2)))  __bf16 v2bf;
typedef __attribute__((ext_vector_type(8)))  float  v8f;

#if defined(__has_builtin)
#if __has_builtin(__builtin_amdgcn_cvt_pk_bf16_f32)
#define HAVE_CVT_PK_BF16 1
#endif
#endif

// pack two fp32 -> two bf16 in one dword (RNE)
__device__ inline unsigned int pack2bf(float a, float b) {
#ifdef HAVE_CVT_PK_BF16
    v2bf p = __builtin_amdgcn_cvt_pk_bf16_f32(a, b);
    unsigned int u;
    __builtin_memcpy(&u, &p, 4);
    return u;
#else
    unsigned int ua = __float_as_uint(a);
    ua = ua + 0x7FFFu + ((ua >> 16) & 1u);
    unsigned int ub = __float_as_uint(b);
    ub = ub + 0x7FFFu + ((ub >> 16) & 1u);
    return (ua >> 16) | (ub & 0xFFFF0000u);
#endif
}

__device__ inline bf16_t f2bf(float f) {
    unsigned int u = pack2bf(f, 0.0f);
    unsigned short h = (unsigned short)u;
    bf16_t b;
    __builtin_memcpy(&b, &h, 2);
    return b;
}

// ---------------------------------------------------------------------------
// WMMA bf16 GEMM: C[M,N] = A(M,K) * Bm(K,N) + bias(N)
// LAYOUT == 0 : A row-major (lda = K)
// LAYOUT == 1 : A is z stored (B, K, L); A[m][k] = z[((m/L)*K + k)*L + m%L]
//               (contiguous along m within a 128-row tile; L % TILE_M == 0)
// Block tile 128x128, K-step 32, 256 threads = 8 waves (4x2), wave tile 32x64.
// ---------------------------------------------------------------------------
#define TILE_M 128
#define TILE_N 128
#define TILE_K 32
#define LDT 40   // padded LDS stride (bf16 elements); rows stay 8B aligned

template <int LAYOUT>
__global__ __launch_bounds__(256)
void gemm_bf16_wmma(const float* __restrict__ A, const float* __restrict__ Bm,
                    const float* __restrict__ bias, float* __restrict__ C,
                    int M, int N, int K)
{
    __shared__ bf16_t As[TILE_M * LDT];
    __shared__ bf16_t Bs[TILE_N * LDT];   // stored transposed: [n][k]

    const int tid  = threadIdx.x;
    const int lane = tid & 31;
    const int wid  = tid >> 5;
    const int wm   = wid >> 1;   // 0..3 -> 32-row band
    const int wn   = wid & 1;    // 0..1 -> 64-col band

    const int m0 = blockIdx.y * TILE_M;
    const int n0 = blockIdx.x * TILE_N;

    v8f acc[2][4];
#pragma unroll
    for (int i = 0; i < 2; ++i)
#pragma unroll
        for (int j = 0; j < 4; ++j)
#pragma unroll
            for (int r = 0; r < 8; ++r) acc[i][j][r] = 0.0f;

    for (int k0 = 0; k0 < K; k0 += TILE_K) {
        // ---- stage A tile (fp32 -> bf16), exact trip counts, b128 loads ----
        if (LAYOUT == 0) {
            // contiguous along k: float4 load + packed b64 LDS store
#pragma unroll
            for (int it = 0; it < (TILE_M * TILE_K / 4) / 256; ++it) {
                const int idx = tid + it * 256;
                const int r  = idx >> 3;           // 8 quads per 32-wide row
                const int cq = (idx & 7) * 4;
                const float4 v = *(const float4*)&A[(size_t)(m0 + r) * K + k0 + cq];
                uint2 p;
                p.x = pack2bf(v.x, v.y);
                p.y = pack2bf(v.z, v.w);
                *(uint2*)&As[r * LDT + cq] = p;
            }
        } else {
            // z layout: contiguous along m (== t); float4 along rows
#pragma unroll
            for (int it = 0; it < ((TILE_M / 4) * TILE_K) / 256; ++it) {
                const int idx = tid + it * 256;
                const int c  = idx & 31;
                const int rq = (idx >> 5) * 4;
                const int m  = m0 + rq;
                const int b  = m / Lc;
                const int t  = m - b * Lc;
                const float4 v =
                    *(const float4*)&A[((size_t)(b * K + k0 + c)) * Lc + t];
                As[(rq + 0) * LDT + c] = f2bf(v.x);
                As[(rq + 1) * LDT + c] = f2bf(v.y);
                As[(rq + 2) * LDT + c] = f2bf(v.z);
                As[(rq + 3) * LDT + c] = f2bf(v.w);
            }
        }
        // ---- stage B tile transposed; float4 along n, scalar bf16 stores ----
#pragma unroll
        for (int it = 0; it < (TILE_K * TILE_N / 4) / 256; ++it) {
            const int idx = tid + it * 256;
            const int k  = idx >> 5;               // 32 quads per 128-wide row
            const int nq = (idx & 31) * 4;
            const float4 v = *(const float4*)&Bm[(size_t)(k0 + k) * N + n0 + nq];
            Bs[(nq + 0) * LDT + k] = f2bf(v.x);
            Bs[(nq + 1) * LDT + k] = f2bf(v.y);
            Bs[(nq + 2) * LDT + k] = f2bf(v.z);
            Bs[(nq + 3) * LDT + k] = f2bf(v.w);
        }
        if (k0 + TILE_K < K)  // hint next B tile (global_prefetch_b8)
            __builtin_prefetch(&Bm[(size_t)(k0 + TILE_K) * N + n0 + (tid & 127)], 0, 3);
        __syncthreads();

        // fragment loads per CDNA5 16-bit operand layout:
        // lanes 0-15: K = {0..7, 16..23}; lanes 16-31: K = {8..15, 24..31}
        const int row   = lane & 15;
        const int khalf = (lane >> 4) * 8;

        v16bf afr[2], bfr[4];
#pragma unroll
        for (int i = 0; i < 2; ++i) {
            const bf16_t* p = &As[(wm * 32 + i * 16 + row) * LDT];
#pragma unroll
            for (int e = 0; e < 8; ++e) {
                afr[i][e]     = p[khalf + e];
                afr[i][8 + e] = p[16 + khalf + e];
            }
        }
#pragma unroll
        for (int j = 0; j < 4; ++j) {
            const bf16_t* p = &Bs[(wn * 64 + j * 16 + row) * LDT];
#pragma unroll
            for (int e = 0; e < 8; ++e) {
                bfr[j][e]     = p[khalf + e];
                bfr[j][8 + e] = p[16 + khalf + e];
            }
        }
#pragma unroll
        for (int i = 0; i < 2; ++i)
#pragma unroll
            for (int j = 0; j < 4; ++j)
                acc[i][j] = __builtin_amdgcn_wmma_f32_16x16x32_bf16(
                    false, afr[i], false, bfr[j], (short)0, acc[i][j], false, false);
        __syncthreads();
    }

    // D layout: VGPR r -> M = r + 8*(lane>=16), N = lane&15
    const int lr = (lane >> 4) * 8;
    const int lc = lane & 15;
#pragma unroll
    for (int i = 0; i < 2; ++i)
#pragma unroll
        for (int j = 0; j < 4; ++j) {
            int gm = m0 + wm * 32 + i * 16 + lr;
            int gn = n0 + wn * 64 + j * 16 + lc;
            float bv = bias[gn];
#pragma unroll
            for (int r = 0; r < 8; ++r)
                C[(size_t)(gm + r) * N + gn] = acc[i][j][r] + bv;
        }
}

// ---------------------------------------------------------------------------
// Depthwise causal conv (k=3) + split + gate, 4 channels per thread so every
// read (taps, weights, biases) is an aligned b128 load from contiguous rows.
//   x0 = conv(ch d), x1 = conv(ch D+d), v = conv(ch 2D+d); vg = v*x1
// up is (B, L, 3D) row-major; outputs are (B, D, L).
// ---------------------------------------------------------------------------
__global__ __launch_bounds__(256)
void conv_split(const float* __restrict__ up, const float* __restrict__ sf_w,
                const float* __restrict__ sf_b, float* __restrict__ x0,
                float* __restrict__ vg)
{
    int n = blockIdx.x * blockDim.x + threadIdx.x;
    if (n >= Bc * Lc * (Dc / 4)) return;
    const int dq = (n & (Dc / 4 - 1)) * 4;   // channel quad
    const int t  = (n >> 8) & (Lc - 1);
    const int b  = n >> 20;

    const float* row = up + ((size_t)b * Lc + t) * TDc;
    float4 res[3];
#pragma unroll
    for (int s = 0; s < 3; ++s) {
        const int c = s * Dc + dq;
        const float4 cur = *(const float4*)&row[c];
        float4 p1 = make_float4(0.f, 0.f, 0.f, 0.f);
        float4 p2 = make_float4(0.f, 0.f, 0.f, 0.f);
        if (t >= 1) p1 = *(const float4*)&row[c - TDc];
        if (t >= 2) p2 = *(const float4*)&row[c - 2 * TDc];
        // sf_w rows for channels c..c+3 are 12 contiguous floats
        const float4 w0 = *(const float4*)&sf_w[c * 3 + 0];
        const float4 w1 = *(const float4*)&sf_w[c * 3 + 4];
        const float4 w2 = *(const float4*)&sf_w[c * 3 + 8];
        const float4 bb = *(const float4*)&sf_b[c];
        float4 r;
        r.x = bb.x + w0.x * p2.x + w0.y * p1.x + w0.z * cur.x;
        r.y = bb.y + w0.w * p2.y + w1.x * p1.y + w1.y * cur.y;
        r.z = bb.z + w1.z * p2.z + w1.w * p1.z + w2.x * cur.z;
        r.w = bb.w + w2.y * p2.w + w2.z * p1.w + w2.w * cur.w;
        res[s] = r;
    }
#pragma unroll
    for (int i = 0; i < 4; ++i) {
        const size_t o = ((size_t)b * Dc + dq + i) * Lc + t;
        const float a0 = (&res[0].x)[i];
        const float a1 = (&res[1].x)[i];
        const float av = (&res[2].x)[i];
        x0[o] = a0;
        vg[o] = av * a1;
    }
}

// ---------------------------------------------------------------------------
// Implicit filter: k(t, d) = modulated SIREN MLP. One block per t (64 threads).
// ---------------------------------------------------------------------------
__global__ __launch_bounds__(64)
void filter_k(const float* __restrict__ f_w1, const float* __restrict__ f_b1,
              const float* __restrict__ freq1,
              const float* __restrict__ f_w2, const float* __restrict__ f_b2,
              const float* __restrict__ freq2,
              const float* __restrict__ f_w3, const float* __restrict__ f_b3,
              const float* __restrict__ freq3,
              const float* __restrict__ f_wout, float* __restrict__ kbuf)
{
    __shared__ float h[2][FO];
    const int t = blockIdx.x;
    const int j = threadIdx.x;

    const float t01 = (float)t / (float)(Lc - 1);
    const float w   = 6.283185307179586f * (float)t / (float)Lc;
    const float fw  = 1e-4f * w;            // bands == 1, f = 1e-4
    const float z0 = t01, z1 = cosf(fw), z2 = -sinf(fw);

    float a = z0 * f_w1[j] + z1 * f_w1[FO + j] + z2 * f_w1[2 * FO + j] + f_b1[j];
    h[0][j] = sinf(freq1[j] * a);
    __syncthreads();

    float s = f_b2[j];
    for (int i = 0; i < FO; ++i) s += h[0][i] * f_w2[i * FO + j];
    h[1][j] = sinf(freq2[j] * s);
    __syncthreads();

    s = f_b3[j];
    for (int i = 0; i < FO; ++i) s += h[1][i] * f_w3[i * FO + j];
    h[0][j] = sinf(freq3[j] * s);
    __syncthreads();

    const float max_decay = -4.605170185988091f / 0.3f;
    const float min_decay = -4.605170185988091f / 1.5f;
    for (int o = 0; o < Dc / FO; ++o) {
        int dch = j + o * FO;
        float acc = 0.0f;
        for (int i = 0; i < FO; ++i) acc += h[0][i] * f_wout[i * Dc + dch];
        float delta = min_decay + (max_decay - min_decay) * (float)dch / (float)(Dc - 1);
        float decay = __expf(-t01 * fabsf(delta));
        kbuf[(size_t)t * Dc + dch] = acc * (decay + 0.05f);
    }
}

// ---------------------------------------------------------------------------
// In-LDS radix-2 FFT, 8192 points, 256 threads, sign=-1 forward / +1 inverse.
// ---------------------------------------------------------------------------
__device__ void fft_lds(float* re, float* im, float sign)
{
    const int tid = threadIdx.x;
    for (int i = tid; i < FFT_N; i += 256) {
        int j = (int)(__brev((unsigned)i) >> (32 - FFT_LOG));
        if (j > i) {
            float tr = re[i]; re[i] = re[j]; re[j] = tr;
            float ti = im[i]; im[i] = im[j]; im[j] = ti;
        }
    }
    __syncthreads();
    for (int s = 1; s <= FFT_LOG; ++s) {
        const int half = 1 << (s - 1);
        const float ang0 = sign * 6.283185307179586f / (float)(1 << s);
        for (int bf = tid; bf < FFT_N / 2; bf += 256) {
            int grp = bf >> (s - 1);
            int pos = bf & (half - 1);
            int i = (grp << s) + pos;
            int j2 = i + half;
            float sn, cs;
            __sincosf(ang0 * (float)pos, &sn, &cs);
            float jr = re[j2], ji = im[j2];
            float tr = cs * jr - sn * ji;
            float ti = cs * ji + sn * jr;
            float ir = re[i], ii = im[i];
            re[j2] = ir - tr; im[j2] = ii - ti;
            re[i]  = ir + tr; im[i]  = ii + ti;
        }
        __syncthreads();
    }
}

__global__ __launch_bounds__(256)
void fft_of_k(const float* __restrict__ kbuf, float2* __restrict__ Khat)
{
    __shared__ float re[FFT_N];
    __shared__ float im[FFT_N];
    const int d = blockIdx.x;
    for (int i = threadIdx.x; i < FFT_N; i += 256) {
        re[i] = (i < Lc) ? kbuf[(size_t)i * Dc + d] : 0.0f;
        im[i] = 0.0f;
    }
    __syncthreads();
    fft_lds(re, im, -1.0f);
    for (int i = threadIdx.x; i < FFT_N; i += 256)
        Khat[(size_t)d * FFT_N + i] = make_float2(re[i], im[i]);
}

// forward FFT of vg row, spectral multiply by Khat[d], inverse FFT, fused
// epilogue: z = (y + vg*filt_bias) * x0, stored (B, D, L).
__global__ __launch_bounds__(256)
void fft_conv(const float* __restrict__ vg, const float* __restrict__ x0,
              const float2* __restrict__ Khat, const float* __restrict__ filt_bias,
              float* __restrict__ zbuf)
{
    __shared__ float re[FFT_N];
    __shared__ float im[FFT_N];
    const int n = blockIdx.x;          // b*D + d
    const int d = n & (Dc - 1);
    const float* vrow = vg + (size_t)n * Lc;

    for (int i = threadIdx.x; i < FFT_N; i += 256) {
        re[i] = (i < Lc) ? vrow[i] : 0.0f;
        im[i] = 0.0f;
    }
    __syncthreads();
    fft_lds(re, im, -1.0f);

    const float2* kr = Khat + (size_t)d * FFT_N;
    for (int i = threadIdx.x; i < FFT_N; i += 256) {
        float2 kv = kr[i];
        float r = re[i], q = im[i];
        re[i] = r * kv.x - q * kv.y;
        im[i] = r * kv.y + q * kv.x;
    }
    __syncthreads();
    fft_lds(re, im, 1.0f);

    const float fb = filt_bias[d];
    const float* x0row = x0 + (size_t)n * Lc;
    float* zrow = zbuf + (size_t)n * Lc;
    const float invN = 1.0f / (float)FFT_N;
    for (int i = threadIdx.x; i < Lc; i += 256) {
        float y = re[i] * invN;
        zrow[i] = (y + vrow[i] * fb) * x0row[i];
    }
}

// ---------------------------------------------------------------------------
extern "C" void kernel_launch(void* const* d_in, const int* in_sizes, int n_in,
                              void* d_out, int out_size, void* d_ws, size_t ws_size,
                              hipStream_t stream)
{
    (void)in_sizes; (void)n_in; (void)out_size; (void)ws_size;
    const float* u         = (const float*)d_in[0];
    const float* w_in      = (const float*)d_in[1];
    const float* b_in      = (const float*)d_in[2];
    const float* sf_w      = (const float*)d_in[3];
    const float* sf_b      = (const float*)d_in[4];
    const float* f_w1      = (const float*)d_in[5];
    const float* f_b1      = (const float*)d_in[6];
    const float* freq1     = (const float*)d_in[7];
    const float* f_w2      = (const float*)d_in[8];
    const float* f_b2      = (const float*)d_in[9];
    const float* freq2     = (const float*)d_in[10];
    const float* f_w3      = (const float*)d_in[11];
    const float* f_b3      = (const float*)d_in[12];
    const float* freq3     = (const float*)d_in[13];
    const float* f_wout    = (const float*)d_in[14];
    const float* filt_bias = (const float*)d_in[15];
    const float* w_out     = (const float*)d_in[16];
    const float* b_out     = (const float*)d_in[17];

    char* ws = (char*)d_ws;
    size_t off = 0;
    auto carve = [&](size_t bytes) {
        void* p = ws + off;
        off += (bytes + 255) & ~(size_t)255;
        return p;
    };
    float*  up   = (float*) carve((size_t)Bc * Lc * TDc * 4);   // 384 MB
    float*  x0   = (float*) carve((size_t)Bc * Dc * Lc * 4);    // 128 MB
    float*  vg   = (float*) carve((size_t)Bc * Dc * Lc * 4);    // 128 MB
    float*  kbuf = (float*) carve((size_t)Lc * Dc * 4);         //  16 MB
    float2* Khat = (float2*)carve((size_t)Dc * FFT_N * 8);      //  64 MB
    float*  zbuf = (float*) carve((size_t)Bc * Dc * Lc * 4);    // 128 MB

    const int M = Bc * Lc;  // 32768

    // 1) up = u @ w_in + b_in    (M x 3D, K = D) — WMMA, row-major A
    dim3 g1(TDc / TILE_N, M / TILE_M);
    gemm_bf16_wmma<0><<<g1, 256, 0, stream>>>(u, w_in, b_in, up, M, TDc, Dc);

    // 2) depthwise causal conv + split + gate (vectorized channel quads)
    int tot = Bc * Lc * (Dc / 4);
    conv_split<<<(tot + 255) / 256, 256, 0, stream>>>(up, sf_w, sf_b, x0, vg);

    // 3) implicit filter k(t,d)
    filter_k<<<Lc, 64, 0, stream>>>(f_w1, f_b1, freq1, f_w2, f_b2, freq2,
                                    f_w3, f_b3, freq3, f_wout, kbuf);

    // 4) K̂ = FFT(k) per channel
    fft_of_k<<<Dc, 256, 0, stream>>>(kbuf, Khat);

    // 5) per-(b,d) FFT conv with fused gating epilogue -> z (B,D,L)
    fft_conv<<<Bc * Dc, 256, 0, stream>>>(vg, x0, Khat, filt_bias, zbuf);

    // 6) out = z @ w_out + b_out  (z gathered (b,d,t)->(m,k)) — WMMA
    dim3 g2(Dc / TILE_N, M / TILE_M);
    gemm_bf16_wmma<1><<<g2, 256, 0, stream>>>(zbuf, w_out, b_out, (float*)d_out,
                                              M, Dc, Dc);
}
